// StandardTransformer_9337258902282
// MI455X (gfx1250) — matrix operand order repaired and checked by reading, not compiled, not run on hardware
//
#include <hip/hip_runtime.h>

// ---------------------------------------------------------------------------
// StandardTransformer forward for MI455X (gfx1250, wave32, WMMA).
// GEMMs: bf16 inputs, f32 accumulate via v_wmma_f32_16x16x32_bf16, fed from a
// double-buffered LDS pipeline (64-deep K stages) filled with
// GLOBAL_LOAD_ASYNC_TO_LDS_B128 (ASYNCcnt-tracked CDNA5 async copies).
// ---------------------------------------------------------------------------

typedef __bf16 v8bf  __attribute__((ext_vector_type(8)));
typedef __bf16 v16bf __attribute__((ext_vector_type(16)));
typedef float  v8f   __attribute__((ext_vector_type(8)));

// ---------------------------- constants ------------------------------------
static constexpr int Vv = 32000, Dd = 512, Hh = 8, Ll = 6, Ff = 2048, Ss = 1024, Bb = 2;
static constexpr int DKk = Dd / Hh;           // 64
static constexpr int Mm  = Bb * Ss;           // 2048

// -------------------- CDNA5 async global->LDS helpers ----------------------
__device__ __forceinline__ void async_copy16(const void* gaddr, unsigned lds_addr)
{
    asm volatile("global_load_async_to_lds_b128 %0, %1, off"
                 :: "v"(lds_addr), "v"((unsigned long long)(uintptr_t)gaddr)
                 : "memory");
}
__device__ __forceinline__ void wait_async()
{
    asm volatile("s_wait_asynccnt 0x0" ::: "memory");
}

// ---------------------------- GEMM (WMMA) ----------------------------------
// C[M,N] = act( scale * (A[M,K] @ Bt[N,K]^T) + bias[N] )
// A: bf16 row-major (ld = lda), Bt: bf16 K-major (ld = ldb), C ld = ldc.
// Batched over grid.z with (zb, zh) = (z/HH, z%HH) strides for A, B, C.
// K must be a multiple of 64 (stage depth).
template <int WM, int WN, int NW, bool RELU, bool SF32, bool SBF16>
__global__ __launch_bounds__(WM * WN * 32)
void gemm_bf16_wmma(const __bf16* __restrict__ A, const __bf16* __restrict__ Bt,
                    const float* __restrict__ bias,
                    float* __restrict__ Cf, __bf16* __restrict__ Cb,
                    int K, int lda, int ldb, int ldc, float scale,
                    int HH, long sAb, long sAh, long sBb, long sBh,
                    long sCb, long sCh)
{
    constexpr int THREADS = WM * WN * 32;
    constexpr int AROWS = WM * 16;          // rows staged per block
    constexpr int BCOLS = WN * NW * 16;     // cols staged per block
    constexpr int KS    = 64;               // K halves per stage
    constexpr int CHK   = KS / 8;           // 16B chunks per row (8)
    constexpr int LDSW  = KS + 8;           // +8 pad halves (144B rows)
    constexpr int AIT   = AROWS * CHK / THREADS;   // exact trip counts
    constexpr int BIT   = BCOLS * CHK / THREADS;

    __shared__ __align__(64) __bf16 Ash[2][AROWS][LDSW];
    __shared__ __align__(64) __bf16 Bsh[2][BCOLS][LDSW];

    const int z  = blockIdx.z;
    const int zb = z / HH, zh = z % HH;
    A  += (size_t)zb * sAb + (size_t)zh * sAh;
    Bt += (size_t)zb * sBb + (size_t)zh * sBh;
    const size_t coff = (size_t)zb * sCb + (size_t)zh * sCh;

    const int tid  = threadIdx.x;
    const int wave = tid >> 5;
    const int lane = tid & 31;
    const int wm = wave % WM, wn = wave / WM;
    const int rowBase = blockIdx.y * AROWS;
    const int colBase = blockIdx.x * BCOLS;

    // A layout (16-bit A 16x32): lane&15 = row; lane half picks K chunks
    const int am   = lane & 15;
    const int aoff = (lane & 16) ? 8 : 0;
    // B layout (16-bit B 32x16): lane&15 = col; lane half picks K chunks
    const int bn   = lane & 15;
    const int koff = (lane & 16) ? 8 : 0;

    v8f acc[NW];
#pragma unroll
    for (int j = 0; j < NW; ++j)
#pragma unroll
        for (int r = 0; r < 8; ++r) acc[j][r] = 0.0f;

    // stage k-panel [k0, k0+KS) into LDS buffer `buf` (async, 16B chunks)
    auto copy_stage = [&](int k0, int buf) {
#pragma unroll
        for (int i = 0; i < AIT; ++i) {
            const int c  = tid + i * THREADS;
            const int r  = c / CHK;
            const int o8 = (c % CHK) * 8;
            const __bf16* g = A + (size_t)(rowBase + r) * lda + k0 + o8;
            async_copy16(g, (unsigned)(uintptr_t)&Ash[buf][r][o8]);
        }
#pragma unroll
        for (int i = 0; i < BIT; ++i) {
            const int c   = tid + i * THREADS;
            const int col = c / CHK;
            const int o8  = (c % CHK) * 8;
            const __bf16* g = Bt + (size_t)(colBase + col) * ldb + k0 + o8;
            async_copy16(g, (unsigned)(uintptr_t)&Bsh[buf][col][o8]);
        }
    };

    const int nsteps = K / KS;

    copy_stage(0, 0);
    wait_async();
    __syncthreads();

    for (int s = 0; s < nsteps; ++s) {
        const int buf = s & 1;
        if (s + 1 < nsteps) copy_stage((s + 1) * KS, buf ^ 1);

#pragma unroll
        for (int ks = 0; ks < KS; ks += 32) {
            // A fragment from LDS (two aligned 16B reads -> v16bf)
            const __bf16* a0 = &Ash[buf][wm * 16 + am][ks + aoff];
            v8bf alo = *(const v8bf*)a0;
            v8bf ahi = *(const v8bf*)(a0 + 16);
            v16bf a  = __builtin_shufflevector(alo, ahi,
                           0, 1, 2, 3, 4, 5, 6, 7, 8, 9, 10, 11, 12, 13, 14, 15);
#pragma unroll
            for (int j = 0; j < NW; ++j) {
                const __bf16* b0 = &Bsh[buf][wn * (NW * 16) + j * 16 + bn][ks + koff];
                v8bf blo = *(const v8bf*)b0;
                v8bf bhi = *(const v8bf*)(b0 + 16);
                v16bf b  = __builtin_shufflevector(blo, bhi,
                               0, 1, 2, 3, 4, 5, 6, 7, 8, 9, 10, 11, 12, 13, 14, 15);
                acc[j] = __builtin_amdgcn_wmma_f32_16x16x32_bf16(
                             false, a, false, b, (short)0, acc[j], false, false);
            }
        }

        wait_async();       // copies into buf^1 have landed
        __syncthreads();    // everyone done reading buf / writing buf^1
    }

    // C/D layout: lane&15 = col; VGPR r -> row r (+8 for upper lane half)
    const int cn = lane & 15;
    const int cm = (lane & 16) ? 8 : 0;
#pragma unroll
    for (int j = 0; j < NW; ++j) {
        const int col = colBase + wn * (NW * 16) + j * 16 + cn;
        const float bv = bias ? bias[col] : 0.0f;
#pragma unroll
        for (int r = 0; r < 8; ++r) {
            const int row = rowBase + wm * 16 + cm + r;
            float v = acc[j][r] * scale + bv;
            if (RELU) v = fmaxf(v, 0.0f);
            const size_t idx = coff + (size_t)row * ldc + col;
            if (SF32)  Cf[idx] = v;
            if (SBF16) Cb[idx] = (__bf16)v;
        }
    }
}

// ------------------------- support kernels ---------------------------------
__global__ void embed_pe_kernel(const int* __restrict__ ids,
                                const float* __restrict__ emb,
                                float* __restrict__ X, __bf16* __restrict__ Xb)
{
    const int row = blockIdx.x;           // B*S rows
    const int s   = row % Ss;
    const int tok = ids[row];
    const float sc = sqrtf((float)Dd);
    const float kf = -9.210340371976184f / (float)Dd;   // -ln(10000)/D
    for (int c = threadIdx.x; c < Dd; c += blockDim.x) {
        const int   i2  = c & ~1;
        const float arg = (float)s * __expf(kf * (float)i2);
        const float pe  = (c & 1) ? __cosf(arg) : __sinf(arg);
        const float v   = emb[(size_t)tok * Dd + c] * sc + pe;
        X [(size_t)row * Dd + c] = v;
        Xb[(size_t)row * Dd + c] = (__bf16)v;
    }
}

__global__ void transpose_cvt_kernel(const float* __restrict__ W,
                                     __bf16* __restrict__ Wt, int K, int N)
{
    const size_t idx = (size_t)blockIdx.x * blockDim.x + threadIdx.x;
    if (idx >= (size_t)K * N) return;
    const int n = (int)(idx / K);
    const int k = (int)(idx % K);
    Wt[idx] = (__bf16)W[(size_t)k * N + n];
}

// V [B,S,H,DK] bf16 -> Vt [B,H,DK,S] bf16 (K-major for the PV GEMM)
__global__ void repack_vt_kernel(const __bf16* __restrict__ Vsrc,
                                 __bf16* __restrict__ Vt)
{
    const size_t idx = (size_t)blockIdx.x * blockDim.x + threadIdx.x;
    const int s = (int)(idx % Ss);
    size_t t = idx / Ss;
    const int d = (int)(t % DKk); t /= DKk;
    const int h = (int)(t % Hh);
    const int b = (int)(t / Hh);
    Vt[idx] = Vsrc[(((size_t)b * Ss + s) * Hh + h) * DKk + d];
}

__global__ void softmax_causal_kernel(const float* __restrict__ scores,
                                      __bf16* __restrict__ P)
{
    __shared__ float red[256];
    const int q = blockIdx.x;
    const size_t base = ((size_t)blockIdx.y * Ss + q) * Ss;
    const int tid = threadIdx.x;

    float mx = -3.0e38f;
    for (int j = tid; j <= q; j += 256) mx = fmaxf(mx, scores[base + j]);
    red[tid] = mx; __syncthreads();
    for (int s = 128; s > 0; s >>= 1) {
        if (tid < s) red[tid] = fmaxf(red[tid], red[tid + s]);
        __syncthreads();
    }
    mx = red[0]; __syncthreads();

    float sum = 0.0f;
    for (int j = tid; j <= q; j += 256) sum += __expf(scores[base + j] - mx);
    red[tid] = sum; __syncthreads();
    for (int s = 128; s > 0; s >>= 1) {
        if (tid < s) red[tid] += red[tid + s];
        __syncthreads();
    }
    const float inv = 1.0f / red[0];

    for (int j = tid; j <= q; j += 256)
        P[base + j] = (__bf16)(__expf(scores[base + j] - mx) * inv);
    for (int j = q + 1 + tid; j < Ss; j += 256)
        P[base + j] = (__bf16)0.0f;
}

// y = LN(x (+ r)) * g + b ; writes f32 and bf16 copies. One block per row.
__global__ void add_layernorm_kernel(const float* __restrict__ x,
                                     const float* __restrict__ r,
                                     const float* __restrict__ g,
                                     const float* __restrict__ b,
                                     float* __restrict__ yf,
                                     __bf16* __restrict__ yb)
{
    __shared__ float s1[256], s2[256];
    const int row = blockIdx.x, tid = threadIdx.x;
    constexpr int NPT = Dd / 256;       // 2
    float v[NPT];
    float sum = 0.0f, sq = 0.0f;
#pragma unroll
    for (int i = 0; i < NPT; ++i) {
        const int c = tid + i * 256;
        float t = x[(size_t)row * Dd + c];
        if (r) t += r[(size_t)row * Dd + c];
        v[i] = t; sum += t; sq += t * t;
    }
    s1[tid] = sum; s2[tid] = sq; __syncthreads();
    for (int s = 128; s > 0; s >>= 1) {
        if (tid < s) { s1[tid] += s1[tid + s]; s2[tid] += s2[tid + s]; }
        __syncthreads();
    }
    const float mean = s1[0] / (float)Dd;
    const float var  = s2[0] / (float)Dd - mean * mean;
    const float rst  = rsqrtf(var + 1e-5f);
#pragma unroll
    for (int i = 0; i < NPT; ++i) {
        const int c = tid + i * 256;
        const float y = (v[i] - mean) * rst * g[c] + b[c];
        yf[(size_t)row * Dd + c] = y;
        yb[(size_t)row * Dd + c] = (__bf16)y;
    }
}

// ------------------------------ host ---------------------------------------
extern "C" void kernel_launch(void* const* d_in, const int* in_sizes, int n_in,
                              void* d_out, int out_size, void* d_ws, size_t ws_size,
                              hipStream_t stream)
{
    (void)in_sizes; (void)n_in; (void)out_size; (void)ws_size;

    const int*   ids   = (const int*)  d_in[0];
    const float* emb   = (const float*)d_in[1];
    const float* Wq    = (const float*)d_in[2];
    const float* bq    = (const float*)d_in[3];
    const float* Wk    = (const float*)d_in[4];
    const float* bk    = (const float*)d_in[5];
    const float* Wv    = (const float*)d_in[6];
    const float* bv    = (const float*)d_in[7];
    const float* Wo    = (const float*)d_in[8];
    const float* bo    = (const float*)d_in[9];
    const float* ln1g  = (const float*)d_in[10];
    const float* ln1b  = (const float*)d_in[11];
    const float* ln2g  = (const float*)d_in[12];
    const float* ln2b  = (const float*)d_in[13];
    const float* W1    = (const float*)d_in[14];
    const float* b1    = (const float*)d_in[15];
    const float* W2    = (const float*)d_in[16];
    const float* b2    = (const float*)d_in[17];
    const float* normg = (const float*)d_in[18];
    const float* normb = (const float*)d_in[19];
    const float* Wout  = (const float*)d_in[20];
    const float* bout  = (const float*)d_in[21];
    float* out = (float*)d_out;

    // ---- workspace carving (256B aligned) ----
    char* ws = (char*)d_ws;
    size_t off = 0;
    auto alloc = [&](size_t bytes) -> void* {
        void* p = ws + off;
        off = (off + bytes + 255) & ~((size_t)255);
        return p;
    };
    __bf16* wtq   = (__bf16*)alloc((size_t)Ll * Dd * Dd * 2);
    __bf16* wtk   = (__bf16*)alloc((size_t)Ll * Dd * Dd * 2);
    __bf16* wtv   = (__bf16*)alloc((size_t)Ll * Dd * Dd * 2);
    __bf16* wto   = (__bf16*)alloc((size_t)Ll * Dd * Dd * 2);
    __bf16* wt1   = (__bf16*)alloc((size_t)Ll * Dd * Ff * 2);
    __bf16* wt2   = (__bf16*)alloc((size_t)Ll * Ff * Dd * 2);
    __bf16* wtout = (__bf16*)alloc((size_t)Vv * Dd * 2);
    float*  X     = (float*) alloc((size_t)Mm * Dd * 4);
    __bf16* Xb    = (__bf16*)alloc((size_t)Mm * Dd * 2);
    __bf16* qb    = (__bf16*)alloc((size_t)Mm * Dd * 2);
    __bf16* kb    = (__bf16*)alloc((size_t)Mm * Dd * 2);
    __bf16* vb    = (__bf16*)alloc((size_t)Mm * Dd * 2);
    __bf16* vtb   = (__bf16*)alloc((size_t)Bb * Hh * DKk * Ss * 2);
    float*  sc    = (float*) alloc((size_t)Bb * Hh * Ss * Ss * 4);
    __bf16* pb    = (__bf16*)alloc((size_t)Bb * Hh * Ss * Ss * 2);
    __bf16* ctxb  = (__bf16*)alloc((size_t)Mm * Dd * 2);
    float*  aout  = (float*) alloc((size_t)Mm * Dd * 4);
    __bf16* ff1b  = (__bf16*)alloc((size_t)Mm * Ff * 2);
    float*  ff2   = (float*) alloc((size_t)Mm * Dd * 4);

    auto tg = [](size_t n) { return dim3((unsigned)((n + 255) / 256)); };

    // ---- weight convert + transpose to bf16 K-major ----
    for (int l = 0; l < Ll; ++l) {
        const size_t od = (size_t)l * Dd * Dd;
        transpose_cvt_kernel<<<tg((size_t)Dd * Dd), 256, 0, stream>>>(Wq + od, wtq + od, Dd, Dd);
        transpose_cvt_kernel<<<tg((size_t)Dd * Dd), 256, 0, stream>>>(Wk + od, wtk + od, Dd, Dd);
        transpose_cvt_kernel<<<tg((size_t)Dd * Dd), 256, 0, stream>>>(Wv + od, wtv + od, Dd, Dd);
        transpose_cvt_kernel<<<tg((size_t)Dd * Dd), 256, 0, stream>>>(Wo + od, wto + od, Dd, Dd);
        transpose_cvt_kernel<<<tg((size_t)Dd * Ff), 256, 0, stream>>>(
            W1 + (size_t)l * Dd * Ff, wt1 + (size_t)l * Dd * Ff, Dd, Ff);
        transpose_cvt_kernel<<<tg((size_t)Ff * Dd), 256, 0, stream>>>(
            W2 + (size_t)l * Ff * Dd, wt2 + (size_t)l * Ff * Dd, Ff, Dd);
    }
    transpose_cvt_kernel<<<tg((size_t)Dd * Vv), 256, 0, stream>>>(Wout, wtout, Dd, Vv);

    // ---- embedding + positional encoding ----
    embed_pe_kernel<<<Mm, 256, 0, stream>>>(ids, emb, X, Xb);

    const long SD  = (long)Ss * Dd;
    const long SS2 = (long)Ss * Ss;

    for (int l = 0; l < Ll; ++l) {
        const size_t od = (size_t)l * Dd * Dd;
        // Q, K, V projections: [2048,512] @ [512,512]
        gemm_bf16_wmma<4, 2, 4, false, false, true><<<dim3(4, 32, 1), 256, 0, stream>>>(
            Xb, wtq + od, bq + l * Dd, nullptr, qb, Dd, Dd, Dd, Dd, 1.0f,
            1, 0, 0, 0, 0, 0, 0);
        gemm_bf16_wmma<4, 2, 4, false, false, true><<<dim3(4, 32, 1), 256, 0, stream>>>(
            Xb, wtk + od, bk + l * Dd, nullptr, kb, Dd, Dd, Dd, Dd, 1.0f,
            1, 0, 0, 0, 0, 0, 0);
        gemm_bf16_wmma<4, 2, 4, false, false, true><<<dim3(4, 32, 1), 256, 0, stream>>>(
            Xb, wtv + od, bv + l * Dd, nullptr, vb, Dd, Dd, Dd, Dd, 1.0f,
            1, 0, 0, 0, 0, 0, 0);
        // V -> [B,H,DK,S]
        repack_vt_kernel<<<tg((size_t)Bb * Hh * DKk * Ss), 256, 0, stream>>>(vb, vtb);

        // scores = Q @ K^T / sqrt(DK), batched over B*H (z), K-dim = DK
        gemm_bf16_wmma<4, 2, 4, false, true, false><<<dim3(8, 16, Bb * Hh), 256, 0, stream>>>(
            qb, kb, nullptr, sc, nullptr, DKk, Dd, Dd, Ss, 0.125f,
            Hh, SD, (long)DKk, SD, (long)DKk, (long)Hh * SS2, SS2);

        softmax_causal_kernel<<<dim3(Ss, Bb * Hh), 256, 0, stream>>>(sc, pb);

        // ctx = P @ V : M=S, N=DK, K=S, batched over B*H; output [B,S,H,DK]
        gemm_bf16_wmma<8, 1, 4, false, false, true><<<dim3(1, 8, Bb * Hh), 256, 0, stream>>>(
            pb, vtb, nullptr, nullptr, ctxb, Ss, Ss, Ss, Dd, 1.0f,
            Hh, (long)Hh * SS2, SS2, (long)Hh * DKk * Ss, (long)DKk * Ss,
            SD, (long)DKk);

        // O projection
        gemm_bf16_wmma<4, 2, 4, false, true, false><<<dim3(4, 32, 1), 256, 0, stream>>>(
            ctxb, wto + od, bo + l * Dd, aout, nullptr, Dd, Dd, Dd, Dd, 1.0f,
            1, 0, 0, 0, 0, 0, 0);
        // x = LN(x + attn_out)
        add_layernorm_kernel<<<Mm, 256, 0, stream>>>(
            aout, X, ln1g + l * Dd, ln1b + l * Dd, X, Xb);

        // FFN: relu(x@w1+b1) @ w2 + b2
        gemm_bf16_wmma<4, 2, 4, true, false, true><<<dim3(16, 32, 1), 256, 0, stream>>>(
            Xb, wt1 + (size_t)l * Dd * Ff, b1 + l * Ff, nullptr, ff1b,
            Dd, Dd, Dd, Ff, 1.0f, 1, 0, 0, 0, 0, 0, 0);
        gemm_bf16_wmma<4, 2, 4, false, true, false><<<dim3(4, 32, 1), 256, 0, stream>>>(
            ff1b, wt2 + (size_t)l * Ff * Dd, b2 + l * Dd, ff2, nullptr,
            Ff, Ff, Ff, Dd, 1.0f, 1, 0, 0, 0, 0, 0, 0);
        // x = LN(x + ff)
        add_layernorm_kernel<<<Mm, 256, 0, stream>>>(
            ff2, X, ln2g + l * Dd, ln2b + l * Dd, X, Xb);
    }

    // final LN (no residual)
    add_layernorm_kernel<<<Mm, 256, 0, stream>>>(X, nullptr, normg, normb, X, Xb);

    // logits = x @ out_w + out_b : [2048,512] @ [512,32000] -> d_out
    gemm_bf16_wmma<4, 2, 4, false, true, false><<<dim3(250, 32, 1), 256, 0, stream>>>(
        Xb, wtout, bout, out, nullptr, Dd, Dd, Dd, Vv, 1.0f,
        1, 0, 0, 0, 0, 0, 0);
}
